// Total_Att_85529978733137
// MI455X (gfx1250) — compile-verified, matrix operand order checked by reading, not run on hardware
//
#include <hip/hip_runtime.h>

typedef __attribute__((ext_vector_type(16))) __bf16 v16bf;
typedef __attribute__((ext_vector_type(8)))  float  v8f;

// ---------- helpers ----------

static __device__ __forceinline__ unsigned short f2bf(float f) {
  unsigned int u = __float_as_uint(f);
  u += 0x7fffu + ((u >> 16) & 1u);          // round-to-nearest-even
  return (unsigned short)(u >> 16);
}

static __device__ __forceinline__ v8f wmma_bf16(v16bf a, v16bf b, v8f c) {
  // D(16x16,f32) = A(16x32,bf16) * B(32x16,bf16) + C
  return __builtin_amdgcn_wmma_f32_16x16x32_bf16(false, a, false, b, (short)0, c,
                                                 false, false);
}

static __device__ __forceinline__ v16bf ldfrag(const unsigned short* p) {
  return *reinterpret_cast<const v16bf*>(p);   // 32B aligned fragment load
}

// Fragment storage: per (tile, k-chunk) block of 32 lanes x 16 bf16 (1 KB),
// element order == in-register WMMA order, so loads are two b128 per lane.
// A/B 16-bit layout: lane l holds row/col (l&15); K index for element e is
// k = e + 8*((e>=8) + (l>=16)).

// ---------- kernel 0: fold Wc*W1*W0 and BatchNorm into M' (64x64) + beta ----------
__global__ void k_prep(const float* __restrict__ W0, const float* __restrict__ W1,
                       const float* __restrict__ Wc,
                       const float* __restrict__ bn_scale, const float* __restrict__ bn_bias,
                       const float* __restrict__ bn_mean, const float* __restrict__ bn_var,
                       float* __restrict__ Mp, float* __restrict__ beta) {
  __shared__ float sA[4096];
  __shared__ float sB[4096];
  __shared__ float sT[4096];
  __shared__ float sInv[64];
  int t = threadIdx.x;
  for (int i = t; i < 4096; i += 256) { sA[i] = W0[i]; sB[i] = W1[i]; }
  if (t < 64) {
    float inv = bn_scale[t] * rsqrtf(bn_var[t] + 1e-5f);
    sInv[t] = inv;
    beta[t] = bn_bias[t] - bn_mean[t] * inv;
  }
  __syncthreads();
  for (int j = 0; j < 16; ++j) {            // T = W1 * W0
    int o = t + 256 * j;
    int c = o >> 6, k = o & 63;
    float acc = 0.f;
    for (int w = 0; w < 64; ++w) acc += sB[c * 64 + w] * sA[w * 64 + k];
    sT[o] = acc;
  }
  __syncthreads();
  for (int i = t; i < 4096; i += 256) sA[i] = Wc[i];
  __syncthreads();
  for (int j = 0; j < 16; ++j) {            // M' = diag(inv) * Wc * T
    int o = t + 256 * j;
    int c = o >> 6, k = o & 63;
    float acc = 0.f;
    for (int w = 0; w < 64; ++w) acc += sA[c * 64 + w] * sT[w * 64 + k];
    Mp[o] = sInv[c] * acc;
  }
}

// ---------- kernel 1: QKV projections, packed straight into WMMA fragment layout ----------
__global__ void k_qkv(const float* __restrict__ x,
                      const float* __restrict__ Wq, const float* __restrict__ Wk,
                      const float* __restrict__ Wv,
                      unsigned short* __restrict__ qfrag, unsigned short* __restrict__ kfrag,
                      float* __restrict__ vraw) {
  __shared__ float sWq[4096], sWk[4096], sWv[4096], sX[4096];
  int b = blockIdx.y;
  int n0 = blockIdx.x * 64;
  int t = threadIdx.x;
  for (int i = t; i < 4096; i += 256) {
    sWq[i] = Wq[i]; sWk[i] = Wk[i]; sWv[i] = Wv[i];
    int ci = i >> 6, nl = i & 63;
    sX[i] = x[(b * 64 + ci) * 4096 + n0 + nl];
  }
  __syncthreads();
  int nl = t & 63;
  int co0 = t >> 6;
  int n = n0 + nl;
  for (int j = 0; j < 16; ++j) {
    int co = co0 + 4 * j;
    float q = 0.f, k = 0.f, v = 0.f;
    for (int ci = 0; ci < 64; ++ci) {
      float xv = sX[ci * 64 + nl];
      q += sWq[co * 64 + ci] * xv;
      k += sWk[co * 64 + ci] * xv;
      v += sWv[co * 64 + ci] * xv;
    }
    // inverse fragment mapping for k-index co (within 64 = 2 chunks of 32)
    int col = co & 31, kc = co >> 5;
    int hi = (col >> 3) & 1;
    int e  = (col & 7) + 8 * ((col >> 4) & 1);
    int lane = (n & 15) + 16 * hi;
    qfrag[((b * 256 + (n >> 4)) * 2 + kc) * 512 + lane * 16 + e] = f2bf(q); // A layout, row = n
    kfrag[((b * 256 + (n >> 4)) * 2 + kc) * 512 + lane * 16 + e] = f2bf(k); // B layout, col = m = n
    vraw[(b * 64 + co) * 4096 + n] = v;
  }
}

// ---------- kernel 2: colsum[m] = sum_n exp(E[n,m]), E tiles via WMMA ----------
__global__ void k_colsum(const unsigned short* __restrict__ qfrag,
                         const unsigned short* __restrict__ kfrag,
                         float* __restrict__ colsum) {
  int b = blockIdx.y;
  int wave = threadIdx.x >> 5;
  int lane = threadIdx.x & 31;
  int mtile = blockIdx.x * 4 + wave;
  v16bf k0 = ldfrag(kfrag + ((b * 256 + mtile) * 2 + 0) * 512 + lane * 16);
  v16bf k1 = ldfrag(kfrag + ((b * 256 + mtile) * 2 + 1) * 512 + lane * 16);
  float partial = 0.f;
  for (int nt = 0; nt < 256; ++nt) {
    v16bf q0 = ldfrag(qfrag + ((b * 256 + nt) * 2 + 0) * 512 + lane * 16);
    v16bf q1 = ldfrag(qfrag + ((b * 256 + nt) * 2 + 1) * 512 + lane * 16);
    v8f acc = {0.f, 0.f, 0.f, 0.f, 0.f, 0.f, 0.f, 0.f};
    acc = wmma_bf16(q0, k0, acc);
    acc = wmma_bf16(q1, k1, acc);
    union { v8f v; float f[8]; } u; u.v = acc;
#pragma unroll
    for (int r = 0; r < 8; ++r) partial += __expf(u.f[r]);
  }
  partial += __shfl_xor(partial, 16, 32);   // combine row halves (wave32)
  if (lane < 16) colsum[b * 4096 + mtile * 16 + lane] = partial;
}

// ---------- kernel 3: Vs fragments = V[c,m]/colsum[m] in B-fragment layout ----------
__global__ void k_vs(const float* __restrict__ vraw, const float* __restrict__ colsum,
                     unsigned short* __restrict__ vsfrag) {
  int mc = blockIdx.x, ct = blockIdx.y, b = blockIdx.z;
  int lane = threadIdx.x;
  int hi = lane >> 4;
  int c = ct * 16 + (lane & 15);
  int base = ((b * 4 + ct) * 128 + mc) * 512 + lane * 16;
#pragma unroll
  for (int e = 0; e < 16; ++e) {
    int m = mc * 32 + e + 8 * ((e >> 3) + hi);
    float cs = colsum[b * 4096 + m] * (1.f + 1e-9f);  // softmax + extra renorm
    vsfrag[base + e] = f2bf(vraw[(b * 64 + c) * 4096 + m] / cs);
  }
}

// ---------- kernel 4: out[n,c] = sum_m exp(E[n,m]) * Vs[m,c]  (flash-style) ----------
__global__ void k_attn_out(const unsigned short* __restrict__ qfrag,
                           const unsigned short* __restrict__ kfrag,
                           const unsigned short* __restrict__ vsfrag,
                           float* __restrict__ outb) {
  __shared__ alignas(32) unsigned short sP[4 * 512];  // per-wave 16x32 bf16 P tile
  int b = blockIdx.y;
  int wave = threadIdx.x >> 5;
  int lane = threadIdx.x & 31;
  int ntile = blockIdx.x * 4 + wave;
  unsigned short* myP = sP + wave * 512;
  v16bf q0 = ldfrag(qfrag + ((b * 256 + ntile) * 2 + 0) * 512 + lane * 16);
  v16bf q1 = ldfrag(qfrag + ((b * 256 + ntile) * 2 + 1) * 512 + lane * 16);
  v8f z = {0.f, 0.f, 0.f, 0.f, 0.f, 0.f, 0.f, 0.f};
  v8f acc0 = z, acc1 = z, acc2 = z, acc3 = z;
  int hiL = lane >> 4, col = lane & 15;
  for (int mc = 0; mc < 128; ++mc) {
    int mt0 = mc * 2, mt1 = mc * 2 + 1;
    v16bf k00 = ldfrag(kfrag + ((b * 256 + mt0) * 2 + 0) * 512 + lane * 16);
    v16bf k01 = ldfrag(kfrag + ((b * 256 + mt0) * 2 + 1) * 512 + lane * 16);
    v16bf k10 = ldfrag(kfrag + ((b * 256 + mt1) * 2 + 0) * 512 + lane * 16);
    v16bf k11 = ldfrag(kfrag + ((b * 256 + mt1) * 2 + 1) * 512 + lane * 16);
    v8f e0 = wmma_bf16(q1, k01, wmma_bf16(q0, k00, z));  // E tile, m = mt0
    v8f e1 = wmma_bf16(q1, k11, wmma_bf16(q0, k10, z));  // E tile, m = mt1
    union { v8f v; float f[8]; } u0, u1; u0.v = e0; u1.v = e1;
#pragma unroll
    for (int r = 0; r < 8; ++r) {
      int nloc = r + 8 * hiL;
      {  // D-layout (n=nloc, m=col) -> A-fragment slot, m-chunk element 0..15
        int m32 = col;
        int lp = nloc + 16 * ((m32 >> 3) & 1);
        int ep = (m32 & 7) + 8 * ((m32 >> 4) & 1);
        myP[lp * 16 + ep] = f2bf(__expf(u0.f[r]));
      }
      {  // m-chunk element 16..31
        int m32 = 16 + col;
        int lp = nloc + 16 * ((m32 >> 3) & 1);
        int ep = (m32 & 7) + 8 * ((m32 >> 4) & 1);
        myP[lp * 16 + ep] = f2bf(__expf(u1.f[r]));
      }
    }
    // per-wave LDS relayout: wait on CDNA5 DS counter (no cross-wave barrier needed)
    asm volatile("s_wait_dscnt 0x0" ::: "memory");
    v16bf pa = *reinterpret_cast<const v16bf*>(myP + lane * 16);
    v16bf v0 = ldfrag(vsfrag + ((b * 4 + 0) * 128 + mc) * 512 + lane * 16);
    v16bf v1 = ldfrag(vsfrag + ((b * 4 + 1) * 128 + mc) * 512 + lane * 16);
    v16bf v2 = ldfrag(vsfrag + ((b * 4 + 2) * 128 + mc) * 512 + lane * 16);
    v16bf v3 = ldfrag(vsfrag + ((b * 4 + 3) * 128 + mc) * 512 + lane * 16);
    acc0 = wmma_bf16(pa, v0, acc0);
    acc1 = wmma_bf16(pa, v1, acc1);
    acc2 = wmma_bf16(pa, v2, acc2);
    acc3 = wmma_bf16(pa, v3, acc3);
  }
  union { v8f v; float f[8]; } a[4];
  a[0].v = acc0; a[1].v = acc1; a[2].v = acc2; a[3].v = acc3;
#pragma unroll
  for (int ct = 0; ct < 4; ++ct)
#pragma unroll
    for (int r = 0; r < 8; ++r)
      outb[(b * 4096 + ntile * 16 + r + 8 * hiL) * 64 + ct * 16 + col] = a[ct].f[r];
}

// ---------- kernel 5: y = M' * out^T + beta + x ----------
__global__ void k_final(const float* __restrict__ outb, const float* __restrict__ Mp,
                        const float* __restrict__ beta, const float* __restrict__ x,
                        float* __restrict__ y) {
  __shared__ float sM[4096];     // transposed: sM[k*64+c]
  __shared__ float sO[64 * 65];  // [nl][k], padded against bank conflicts
  __shared__ float sB[64];
  int b = blockIdx.y;
  int n0 = blockIdx.x * 64;
  int t = threadIdx.x;
  for (int i = t; i < 4096; i += 256) {
    sM[(i & 63) * 64 + (i >> 6)] = Mp[i];
    sO[(i >> 6) * 65 + (i & 63)] = outb[(b * 4096 + n0 + (i >> 6)) * 64 + (i & 63)];
  }
  if (t < 64) sB[t] = beta[t];
  __syncthreads();
  int c = t & 63;
  int nl0 = t >> 6;
  for (int j = 0; j < 16; ++j) {
    int nl = nl0 + 4 * j;
    float acc = sB[c];
    for (int k = 0; k < 64; ++k) acc += sM[k * 64 + c] * sO[nl * 65 + k];
    int idx = (b * 64 + c) * 4096 + n0 + nl;
    y[idx] = acc + x[idx];
  }
}

// ---------- launch ----------
extern "C" void kernel_launch(void* const* d_in, const int* in_sizes, int n_in,
                              void* d_out, int out_size, void* d_ws, size_t ws_size,
                              hipStream_t stream) {
  (void)in_sizes; (void)n_in; (void)out_size; (void)ws_size;
  const float* x  = (const float*)d_in[0];
  const float* Wq = (const float*)d_in[1];
  const float* Wk = (const float*)d_in[2];
  const float* Wv = (const float*)d_in[3];
  const float* W0 = (const float*)d_in[4];
  const float* W1 = (const float*)d_in[5];
  const float* Wc = (const float*)d_in[6];
  const float* bs = (const float*)d_in[7];
  const float* bb = (const float*)d_in[8];
  const float* bm = (const float*)d_in[9];
  const float* bv = (const float*)d_in[10];

  char* w = (char*)d_ws;
  const size_t MB = (size_t)1 << 20;
  unsigned short* qfrag  = (unsigned short*)(w + 0 * MB);            // 2 MB
  unsigned short* kfrag  = (unsigned short*)(w + 2 * MB);            // 2 MB
  unsigned short* vsfrag = (unsigned short*)(w + 4 * MB);            // 2 MB
  float* vraw   = (float*)(w + 6 * MB);                              // 4 MB
  float* colsum = (float*)(w + 10 * MB);                             // 64 KB
  float* outb   = (float*)(w + 10 * MB + 128 * 1024);                // 4 MB
  float* Mp     = (float*)(w + 14 * MB + 256 * 1024);                // 16 KB
  float* beta   = (float*)(w + 14 * MB + 256 * 1024 + 16 * 1024);    // 256 B

  k_prep<<<1, 256, 0, stream>>>(W0, W1, Wc, bs, bb, bm, bv, Mp, beta);
  k_qkv<<<dim3(64, 4), 256, 0, stream>>>(x, Wq, Wk, Wv, qfrag, kfrag, vraw);
  k_colsum<<<dim3(64, 4), 128, 0, stream>>>(qfrag, kfrag, colsum);
  k_vs<<<dim3(128, 4, 4), 32, 0, stream>>>(vraw, colsum, vsfrag);
  k_attn_out<<<dim3(64, 4), 128, 0, stream>>>(qfrag, kfrag, vsfrag, outb);
  k_final<<<dim3(64, 4), 256, 0, stream>>>(outb, Mp, beta, x, (float*)d_out);
}
